// Attentione_63144609186298
// MI455X (gfx1250) — matrix operand order repaired and checked by reading, not compiled, not run on hardware
//
#include <hip/hip_runtime.h>

// ---------------------------------------------------------------------------
// MI455X (gfx1250) multi-head attention, bf16 WMMA, flash-attention softmax.
// ---------------------------------------------------------------------------

typedef __attribute__((ext_vector_type(16))) __bf16        v16bf;
typedef __attribute__((ext_vector_type(8)))  float         v8f;
typedef __attribute__((ext_vector_type(8)))  unsigned short ushort8;

#define DIM   1024
#define HEADS 8
#define HDIM  128
#define BATCH 4
#define SEQ   2048
#define ROWS  (BATCH * SEQ)          // 8192
#define QKV_N (3 * DIM)              // 3072
// reference multiplies scores by sqrt(head_dim) (replicates the original bug)
#define SCALE 11.313708498984760f

__device__ __forceinline__ unsigned short f32_to_bf16(float f) {
  unsigned int u = __float_as_uint(f);
  u += 0x7fffu + ((u >> 16) & 1u);         // round-to-nearest-even
  return (unsigned short)(u >> 16);
}

union FragU { struct { ushort8 lo, hi; } s; v16bf v; };

// A fragment (16x32 bf16): lane holds row (lane&15).
// lanes 0-15: K 0..7 + 16..23 ; lanes 16-31: K 8..15 + 24..31
__device__ __forceinline__ v16bf load_fragA(const unsigned short* row, int hi) {
  FragU f;
  const unsigned short* p = row + (hi ? 8 : 0);
  f.s.lo = *(const ushort8*)(p);
  f.s.hi = *(const ushort8*)(p + 16);
  return f.v;
}

// B fragment (32x16 bf16): lane holds column (lane&15) = a row of the
// row-major "weight" operand; lanes 0-15 hold K 0..7 + 16..23, lanes 16-31 K 8..15 + 24..31.
__device__ __forceinline__ v16bf load_fragB(const unsigned short* colrow, int hi) {
  FragU f;
  const unsigned short* p = colrow + (hi ? 8 : 0);
  f.s.lo = *(const ushort8*)(p);
  f.s.hi = *(const ushort8*)(p + 16);
  return f.v;
}

__device__ __forceinline__ v8f wmma_bf16(v16bf a, v16bf b, v8f c) {
  return __builtin_amdgcn_wmma_f32_16x16x32_bf16(
      /*neg_a=*/false, a, /*neg_b=*/false, b,
      /*c_mod=*/(short)0, c, /*reuse_a=*/false, /*reuse_b=*/false);
}

// ---------------------------------------------------------------------------
// f32 -> bf16 conversion (grid-stride)
// ---------------------------------------------------------------------------
__global__ void cvt_bf16(const float* __restrict__ in,
                         unsigned short* __restrict__ out, int n) {
  int i = blockIdx.x * blockDim.x + threadIdx.x;
  int stride = gridDim.x * blockDim.x;
  for (; i < n; i += stride) out[i] = f32_to_bf16(in[i]);
}

// ---------------------------------------------------------------------------
// QKV projection: [8192,1024] x [3072,1024]^T + bias.
// Each wave computes a 64x64 tile (4x4 WMMA tiles). Epilogue scatters into
// q[BH][N][128], k[BH][N][128], vT[BH][128][N] as bf16.
// Grid: 768 blocks x 256 threads = 6144 waves = 128 * 48 tiles.
// ---------------------------------------------------------------------------
__global__ __launch_bounds__(256) void qkv_gemm(
    const unsigned short* __restrict__ xb,    // [8192,1024] bf16
    const unsigned short* __restrict__ wb,    // [3072,1024] bf16
    const float*          __restrict__ bias,  // [3072]
    unsigned short* __restrict__ qd,
    unsigned short* __restrict__ kd,
    unsigned short* __restrict__ vT) {
  const int lane = threadIdx.x & 31;
  const int hi   = (lane >> 4) & 1;
  const int col  = lane & 15;
  const int wid  = threadIdx.x >> 5;
  const int w    = blockIdx.x * 8 + wid;
  const int NB   = QKV_N / 64;                // 48
  const int m0   = (w / NB) * 64;
  const int n0   = (w % NB) * 64;

  v8f zero = {};
  v8f acc[4][4];
#pragma unroll
  for (int i = 0; i < 4; ++i)
#pragma unroll
    for (int j = 0; j < 4; ++j) acc[i][j] = zero;

  for (int kk = 0; kk < DIM; kk += 32) {
    v16bf a[4], b[4];
#pragma unroll
    for (int i = 0; i < 4; ++i)
      a[i] = load_fragA(xb + (size_t)(m0 + i * 16 + col) * DIM + kk, hi);
#pragma unroll
    for (int i = 0; i < 4; ++i)
      b[i] = load_fragB(wb + (size_t)(n0 + i * 16 + col) * DIM + kk, hi);
#pragma unroll
    for (int i = 0; i < 4; ++i)
#pragma unroll
      for (int j = 0; j < 4; ++j) acc[i][j] = wmma_bf16(a[i], b[j], acc[i][j]);
  }

#pragma unroll
  for (int jt = 0; jt < 4; ++jt) {
    const int c   = n0 + jt * 16 + col;       // output column in [0,3072)
    const int t   = c / DIM;                  // 0=q 1=k 2=v
    const int rem = c - t * DIM;
    const int h   = rem / HDIM;
    const int d   = rem % HDIM;
    const float bv = bias[c];
#pragma unroll
    for (int it = 0; it < 4; ++it) {
#pragma unroll
      for (int j = 0; j < 8; ++j) {
        const int r  = m0 + it * 16 + j + 8 * hi;
        const int bi = r / SEQ;
        const int n  = r - bi * SEQ;
        const int bh = bi * HEADS + h;
        const unsigned short val = f32_to_bf16(acc[it][jt][j] + bv);
        if (t == 0)      qd[((size_t)bh * SEQ + n) * HDIM + d] = val;
        else if (t == 1) kd[((size_t)bh * SEQ + n) * HDIM + d] = val;
        else             vT[((size_t)bh * HDIM + d) * SEQ + n] = val;
      }
    }
  }
}

// ---------------------------------------------------------------------------
// Flash attention: one wave per 16 query rows of one (b,h).
// Online softmax; q fragments stay resident; probabilities bounce through a
// 1 KB/wave LDS region to convert C-layout -> A-layout in bf16.
// Grid: 512 blocks x 256 threads = 4096 waves = 32 (bh) * 128 (q tiles).
// ---------------------------------------------------------------------------
__global__ __launch_bounds__(256) void flash_attn(
    const unsigned short* __restrict__ qd,   // [BH, SEQ, 128] bf16
    const unsigned short* __restrict__ kd,   // [BH, SEQ, 128] bf16
    const unsigned short* __restrict__ vT,   // [BH, 128, SEQ] bf16
    unsigned short* __restrict__ ctx) {      // [ROWS, 1024]   bf16
  __shared__ unsigned short ldsP[8][16 * 32];

  const int lane = threadIdx.x & 31;
  const int hi   = (lane >> 4) & 1;
  const int col  = lane & 15;
  const int wid  = threadIdx.x >> 5;
  const int w    = blockIdx.x * 8 + wid;
  const int bh   = w >> 7;                   // / (SEQ/16)
  const int qt   = w & 127;

  const unsigned short* kbh = kd + (size_t)bh * SEQ * HDIM;
  const unsigned short* vbh = vT + (size_t)bh * HDIM * SEQ;
  unsigned short* myP = ldsP[wid];

  v16bf qf[4];
#pragma unroll
  for (int kk = 0; kk < 4; ++kk)
    qf[kk] = load_fragA(qd + ((size_t)bh * SEQ + qt * 16 + col) * HDIM + kk * 32, hi);

  v8f zero = {};
  v8f O[8];
#pragma unroll
  for (int t = 0; t < 8; ++t) O[t] = zero;
  float m[8], l[8];
#pragma unroll
  for (int j = 0; j < 8; ++j) { m[j] = -1e30f; l[j] = 0.0f; }

  for (int kt = 0; kt < SEQ; kt += 32) {
    // ---- scores: S[16 x 32] = q(16x128) . k^T, as two 16x16 C tiles ----
    v8f S0 = zero, S1 = zero;
    const unsigned short* kr0 = kbh + (size_t)(kt + col) * HDIM;
    const unsigned short* kr1 = kbh + (size_t)(kt + 16 + col) * HDIM;
#pragma unroll
    for (int kk = 0; kk < 4; ++kk) {
      v16bf b0 = load_fragB(kr0 + kk * 32, hi);
      v16bf b1 = load_fragB(kr1 + kk * 32, hi);
      S0 = wmma_bf16(qf[kk], b0, S0);
      S1 = wmma_bf16(qf[kk], b1, S1);
    }

    // ---- online softmax (row = j + 8*hi lives on 16 lanes sharing hi) ----
#pragma unroll
    for (int j = 0; j < 8; ++j) {
      float s0 = S0[j] * SCALE, s1 = S1[j] * SCALE;
      float mx = fmaxf(s0, s1);
      mx = fmaxf(mx, __shfl_xor(mx, 1, 32));
      mx = fmaxf(mx, __shfl_xor(mx, 2, 32));
      mx = fmaxf(mx, __shfl_xor(mx, 4, 32));
      mx = fmaxf(mx, __shfl_xor(mx, 8, 32));
      const float mnew = fmaxf(m[j], mx);
      const float f    = __expf(m[j] - mnew);
      m[j] = mnew;
      const float p0 = __expf(s0 - mnew);
      const float p1 = __expf(s1 - mnew);
      float ps = p0 + p1;
      ps += __shfl_xor(ps, 1, 32);
      ps += __shfl_xor(ps, 2, 32);
      ps += __shfl_xor(ps, 4, 32);
      ps += __shfl_xor(ps, 8, 32);
      l[j] = l[j] * f + ps;
#pragma unroll
      for (int t = 0; t < 8; ++t) O[t][j] *= f;   // rescale output row
      const int lr = (j + 8 * hi) * 32 + col;     // C-layout -> LDS row-major
      myP[lr]      = f32_to_bf16(p0);
      myP[lr + 16] = f32_to_bf16(p1);
    }

    // ---- O += P(16x32) . V(32x128) ; B frags are rows of vT (contiguous) --
    v16bf pf = load_fragA(myP + col * 32, hi);    // same-wave DS ordering
#pragma unroll
    for (int t = 0; t < 8; ++t) {
      v16bf bv = load_fragB(vbh + (size_t)(t * 16 + col) * SEQ + kt, hi);
      O[t] = wmma_bf16(pf, bv, O[t]);
    }
  }

  // ---- normalize and write context in [B, N, H*Dh] layout (bf16) ----
  float inv[8];
#pragma unroll
  for (int j = 0; j < 8; ++j) inv[j] = 1.0f / l[j];
  const int b = bh >> 3, h = bh & 7;
#pragma unroll
  for (int t = 0; t < 8; ++t)
#pragma unroll
    for (int j = 0; j < 8; ++j) {
      const size_t r = (size_t)b * SEQ + qt * 16 + j + 8 * hi;
      ctx[r * DIM + h * HDIM + t * 16 + col] = f32_to_bf16(O[t][j] * inv[j]);
    }
}

// ---------------------------------------------------------------------------
// Output projection: ctx[8192,1024] x W0[1024,1024]^T + b0 -> f32 out.
// Grid: 256 blocks x 256 threads = 2048 waves = 128 * 16 tiles of 64x64.
// ---------------------------------------------------------------------------
__global__ __launch_bounds__(256) void out_gemm(
    const unsigned short* __restrict__ cb,    // ctx bf16
    const unsigned short* __restrict__ w0b,   // [1024,1024] bf16
    const float*          __restrict__ b0,
    float* __restrict__ out) {
  const int lane = threadIdx.x & 31;
  const int hi   = (lane >> 4) & 1;
  const int col  = lane & 15;
  const int wid  = threadIdx.x >> 5;
  const int w    = blockIdx.x * 8 + wid;
  const int NB   = DIM / 64;                  // 16
  const int m0   = (w / NB) * 64;
  const int n0   = (w % NB) * 64;

  v8f zero = {};
  v8f acc[4][4];
#pragma unroll
  for (int i = 0; i < 4; ++i)
#pragma unroll
    for (int j = 0; j < 4; ++j) acc[i][j] = zero;

  for (int kk = 0; kk < DIM; kk += 32) {
    v16bf a[4], b[4];
#pragma unroll
    for (int i = 0; i < 4; ++i)
      a[i] = load_fragA(cb + (size_t)(m0 + i * 16 + col) * DIM + kk, hi);
#pragma unroll
    for (int i = 0; i < 4; ++i)
      b[i] = load_fragB(w0b + (size_t)(n0 + i * 16 + col) * DIM + kk, hi);
#pragma unroll
    for (int i = 0; i < 4; ++i)
#pragma unroll
      for (int j = 0; j < 4; ++j) acc[i][j] = wmma_bf16(a[i], b[j], acc[i][j]);
  }

#pragma unroll
  for (int jt = 0; jt < 4; ++jt) {
    const int c = n0 + jt * 16 + col;
    const float bv = b0[c];
#pragma unroll
    for (int it = 0; it < 4; ++it)
#pragma unroll
      for (int j = 0; j < 8; ++j)
        out[(size_t)(m0 + it * 16 + j + 8 * hi) * DIM + c] = acc[it][jt][j] + bv;
  }
}

// ---------------------------------------------------------------------------
extern "C" void kernel_launch(void* const* d_in, const int* in_sizes, int n_in,
                              void* d_out, int out_size, void* d_ws, size_t ws_size,
                              hipStream_t stream) {
  const float* x    = (const float*)d_in[0];   // [4,2048,1024]
  const float* Wqkv = (const float*)d_in[1];   // [3072,1024]
  const float* bqkv = (const float*)d_in[2];   // [3072]
  const float* W0   = (const float*)d_in[3];   // [1024,1024]
  const float* b0   = (const float*)d_in[4];   // [1024]
  float* out = (float*)d_out;

  char* ws = (char*)d_ws;
  unsigned short* xb  = (unsigned short*)ws; ws += (size_t)ROWS * DIM * 2;       // 16 MB
  unsigned short* wqb = (unsigned short*)ws; ws += (size_t)QKV_N * DIM * 2;      //  6 MB
  unsigned short* w0b = (unsigned short*)ws; ws += (size_t)DIM * DIM * 2;        //  2 MB
  unsigned short* qd  = (unsigned short*)ws; ws += (size_t)ROWS * DIM * 2;       // 16 MB
  unsigned short* kd  = (unsigned short*)ws; ws += (size_t)ROWS * DIM * 2;       // 16 MB
  unsigned short* vT  = (unsigned short*)ws; ws += (size_t)ROWS * DIM * 2;       // 16 MB
  unsigned short* ctx = (unsigned short*)ws;                                     // 16 MB

  cvt_bf16<<<1024, 256, 0, stream>>>(x, xb, ROWS * DIM);
  cvt_bf16<<<512, 256, 0, stream>>>(Wqkv, wqb, QKV_N * DIM);
  cvt_bf16<<<256, 256, 0, stream>>>(W0, w0b, DIM * DIM);

  qkv_gemm<<<768, 256, 0, stream>>>(xb, wqb, bqkv, qd, kd, vT);   // 6144 waves
  flash_attn<<<512, 256, 0, stream>>>(qd, kd, vT, ctx);           // 4096 waves
  out_gemm<<<256, 256, 0, stream>>>(ctx, w0b, b0, out);           // 2048 waves
}
